// AdMatcher_76605036691514
// MI455X (gfx1250) — compile-verified, matrix operand order checked by reading, not run on hardware
//
#include <hip/hip_runtime.h>
#include <hip/hip_bf16.h>
#include <math.h>

#define BQ   512
#define DM   256
#define NIDX 200000
#define KTOP 128
#define SEG  2048
#define NSEG 98          // ceil(200000/2048)
#define NEGINF (-__builtin_inff())

typedef __attribute__((ext_vector_type(16))) __bf16 v16bf;
typedef __attribute__((ext_vector_type(8)))  float  v8f;

__device__ __forceinline__ unsigned short f2bf(float f) {
  unsigned int u = __float_as_uint(f);
  u += 0x7FFFu + ((u >> 16) & 1u);          // round-to-nearest-even
  return (unsigned short)(u >> 16);
}
__device__ __forceinline__ float bf2f(unsigned short u) {
  return __uint_as_float(((unsigned int)u) << 16);
}

union FragU { uint4 u[2]; v16bf v; };
__device__ __forceinline__ v16bf load_frag16(const unsigned short* p, int ka, int kb) {
  FragU f;
  f.u[0] = *(const uint4*)(p + ka);
  f.u[1] = *(const uint4*)(p + kb);
  return f.v;
}

// ---------------- fp32 -> bf16 conversion ----------------
__global__ void k_f32_to_bf16(const float* __restrict__ src,
                              unsigned short* __restrict__ dst, size_t n) {
  size_t i = blockIdx.x * (size_t)blockDim.x + threadIdx.x;
  size_t stride = gridDim.x * (size_t)blockDim.x;
  for (; i < n; i += stride) dst[i] = f2bf(src[i]);
}

// ---------------- generic bf16 WMMA GEMM:  C[M,N] = A[M,K] * B[N,K]^T + bias ----------------
__global__ void k_gemm_bf16(const unsigned short* __restrict__ A,
                            const unsigned short* __restrict__ Bw,
                            const float* __restrict__ bias,
                            void* __restrict__ Cout,
                            int M, int N, int K, int relu, int outBf16) {
  const int wave = threadIdx.x >> 5, lane = threadIdx.x & 31;
  const int l16 = lane & 15, hi = lane >> 4;
  const int bm = blockIdx.x * 64, bn = blockIdx.y * 64;
  for (int s = wave * 2; s < wave * 2 + 2; ++s) {      // 8 waves x 2 = 16 subtiles (4x4)
    const int mbase = bm + (s >> 2) * 16;
    const int nbase = bn + (s & 3) * 16;
    const unsigned short* arow = A  + (size_t)(mbase + l16) * K;
    const unsigned short* brow = Bw + (size_t)(nbase + l16) * K;
    v8f acc = {};
    for (int k = 0; k < K; k += 32) {
      __builtin_prefetch(arow + k + 64, 0, 1);         // global_prefetch_b8
      __builtin_prefetch(brow + k + 64, 0, 1);
      v16bf a = load_frag16(arow, k + hi * 8, k + 16 + hi * 8);
      v16bf b = load_frag16(brow, k + hi * 8, k + 16 + hi * 8);
      acc = __builtin_amdgcn_wmma_f32_16x16x32_bf16(false, a, false, b,
                                                    (short)0, acc, false, false);
    }
    const int n = nbase + l16;
    const float bv = bias ? bias[n] : 0.0f;
    for (int r = 0; r < 8; ++r) {
      const int m = mbase + hi * 8 + r;
      float v = acc[r] + bv;
      if (relu) v = fmaxf(v, 0.0f);
      if (outBf16) ((unsigned short*)Cout)[(size_t)m * N + n] = f2bf(v);
      else         ((float*)Cout)[(size_t)m * N + n] = v;
    }
  }
}

// ---------------- retrieval: WMMA scores + per-segment top-128 ----------------
__global__ void k_scores_topk(const unsigned short* __restrict__ Qb,
                              const unsigned short* __restrict__ Ib,
                              float* __restrict__ pool_s, int* __restrict__ pool_i) {
  extern __shared__ float smem[];
  float* sc   = smem;                 // [16][SEG]
  float* redv = smem + 16 * SEG;      // [16][16]
  int*   redp = (int*)(redv + 256);   // [16][16]

  const int qt = blockIdx.x;          // query tile 0..31
  const int seg = blockIdx.y;         // 0..NSEG-1
  const int nsegbase = seg * SEG;
  const int wave = threadIdx.x >> 5, lane = threadIdx.x & 31;
  const int l16 = lane & 15, hi = lane >> 4;

  const unsigned short* arow = Qb + (size_t)(qt * 16 + l16) * DM;
  for (int t = wave; t < SEG / 16; t += 8) {           // 128 candidate tiles / 8 waves
    const int nrow = nsegbase + t * 16 + l16;
    const int nclamp = nrow < NIDX ? nrow : NIDX - 1;
    const unsigned short* brow = Ib + (size_t)nclamp * DM;
    v8f acc = {};
    for (int k = 0; k < DM; k += 32) {
      v16bf a = load_frag16(arow, k + hi * 8, k + 16 + hi * 8);
      v16bf b = load_frag16(brow, k + hi * 8, k + 16 + hi * 8);
      acc = __builtin_amdgcn_wmma_f32_16x16x32_bf16(false, a, false, b,
                                                    (short)0, acc, false, false);
    }
    const bool oob = nrow >= NIDX;
    for (int r = 0; r < 8; ++r) {
      const int m = hi * 8 + r;
      sc[m * SEG + t * 16 + l16] = oob ? NEGINF : acc[r];
    }
  }
  __syncthreads();

  // iterative argmax: 16 threads per row, 128 extractions -> sorted descending
  const int row = threadIdx.x >> 4;
  const int tt = threadIdx.x & 15;
  float* rowsc = sc + row * SEG;
  for (int it = 0; it < KTOP; ++it) {
    float best = NEGINF; int bp = 0;
    for (int j = tt; j < SEG; j += 16) {
      float v = rowsc[j];
      if (v > best) { best = v; bp = j; }
    }
    redv[row * 16 + tt] = best; redp[row * 16 + tt] = bp;
    __syncthreads();
    if (tt == 0) {
      float b2 = NEGINF; int p2 = 0;
      for (int u = 0; u < 16; ++u) {
        float v = redv[row * 16 + u];
        if (v > b2) { b2 = v; p2 = redp[row * 16 + u]; }
      }
      const size_t o = (((size_t)(qt * 16 + row)) * NSEG + seg) * KTOP + it;
      pool_s[o] = b2;
      pool_i[o] = nsegbase + p2;
      rowsc[p2] = NEGINF;
    }
    __syncthreads();
  }
}

// ---------------- per-row merge of segment winners -> global top-128 ----------------
__global__ void k_merge(const float* __restrict__ pool_s, const int* __restrict__ pool_i,
                        int* __restrict__ out_idx, float* __restrict__ out_sc) {
  extern __shared__ float smem[];
  const int TOT = NSEG * KTOP;        // 12544
  float* sc   = smem;                 // [TOT]
  float* redv = smem + TOT;           // [256]
  int*   redp = (int*)(redv + 256);   // [256]
  const int row = blockIdx.x;
  const float* ps = pool_s + (size_t)row * TOT;
  const int*   pi = pool_i + (size_t)row * TOT;
  for (int i = threadIdx.x; i < TOT; i += blockDim.x) sc[i] = ps[i];
  __syncthreads();
  for (int it = 0; it < KTOP; ++it) {
    float best = NEGINF; int bp = 0;
    for (int j = threadIdx.x; j < TOT; j += blockDim.x) {
      float v = sc[j];
      if (v > best) { best = v; bp = j; }
    }
    redv[threadIdx.x] = best; redp[threadIdx.x] = bp;
    __syncthreads();
    for (int off = 128; off > 0; off >>= 1) {
      if ((int)threadIdx.x < off && redv[threadIdx.x + off] > redv[threadIdx.x]) {
        redv[threadIdx.x] = redv[threadIdx.x + off];
        redp[threadIdx.x] = redp[threadIdx.x + off];
      }
      __syncthreads();
    }
    if (threadIdx.x == 0) {
      out_sc[(size_t)row * KTOP + it] = redv[0];
      out_idx[(size_t)row * KTOP + it] = pi[redp[0]];
      sc[redp[0]] = NEGINF;
    }
    __syncthreads();
  }
}

// ---------------- gather candidates (fp32 -> bf16) ----------------
__global__ void k_gather(const int* __restrict__ top_idx, const float* __restrict__ index_embs,
                         unsigned short* __restrict__ cand_bf) {
  const int b = blockIdx.x;
  for (int e = threadIdx.x; e < KTOP * DM; e += blockDim.x) {
    const int kk = e >> 8, dd = e & 255;
    const int idx = top_idx[b * KTOP + kk];
    cand_bf[((size_t)b * KTOP + kk) * DM + dd] = f2bf(index_embs[(size_t)idx * DM + dd]);
  }
}

// ---------------- cross attention (1 query token vs 128 keys, 8 heads) ----------------
__global__ void k_attn(const float* __restrict__ qh, const unsigned short* __restrict__ kh,
                       const unsigned short* __restrict__ vh, unsigned short* __restrict__ ctx_bf) {
  __shared__ float attn[8][KTOP];
  const int b = blockIdx.x;
  const int h = threadIdx.x >> 5, lane = threadIdx.x & 31;    // one wave per head
  const float scale = 0.17677669529663687f;                   // 1/sqrt(32)
  const float qv = qh[(size_t)b * DM + h * 32 + lane];
  const unsigned short* kbase = kh + (size_t)b * KTOP * DM + h * 32 + lane;
  for (int k = 0; k < KTOP; ++k) {
    float p = qv * bf2f(kbase[(size_t)k * DM]);
    for (int off = 16; off > 0; off >>= 1) p += __shfl_xor(p, off, 32);
    if (lane == 0) attn[h][k] = p * scale;
  }
  __syncthreads();
  float m = NEGINF;
  for (int k = lane; k < KTOP; k += 32) m = fmaxf(m, attn[h][k]);
  for (int off = 16; off > 0; off >>= 1) m = fmaxf(m, __shfl_xor(m, off, 32));
  float ssum = 0.0f;
  for (int k = lane; k < KTOP; k += 32) {
    float e = __expf(attn[h][k] - m);
    attn[h][k] = e;
    ssum += e;
  }
  for (int off = 16; off > 0; off >>= 1) ssum += __shfl_xor(ssum, off, 32);
  __syncthreads();
  const float inv = 1.0f / ssum;
  const unsigned short* vbase = vh + (size_t)b * KTOP * DM + h * 32 + lane;
  float c = 0.0f;
  for (int k = 0; k < KTOP; ++k) c += attn[h][k] * inv * bf2f(vbase[(size_t)k * DM]);
  ctx_bf[(size_t)b * DM + h * 32 + lane] = f2bf(c);
}

// ---------------- final matvec: rerank = h1 . W_s2 + b ----------------
__global__ void k_final(const unsigned short* __restrict__ h1, const float* __restrict__ W_s2,
                        const float* __restrict__ b_s2, float* __restrict__ rerank) {
  const int b = blockIdx.x * (blockDim.x >> 5) + (threadIdx.x >> 5);
  const int lane = threadIdx.x & 31;
  float s = 0.0f;
  for (int j = lane; j < DM / 2; j += 32) s += bf2f(h1[(size_t)b * (DM / 2) + j]) * W_s2[j];
  for (int off = 16; off > 0; off >>= 1) s += __shfl_xor(s, off, 32);
  if (lane == 0) rerank[b] = s + b_s2[0];
}

extern "C" void kernel_launch(void* const* d_in, const int* in_sizes, int n_in,
                              void* d_out, int out_size, void* d_ws, size_t ws_size,
                              hipStream_t stream) {
  const float* query   = (const float*)d_in[0];
  const float* index   = (const float*)d_in[1];
  const float* W_cq    = (const float*)d_in[2];
  const float* b_cq    = (const float*)d_in[3];
  const float* W_cd    = (const float*)d_in[4];
  const float* b_cd    = (const float*)d_in[5];
  const float* in_w    = (const float*)d_in[6];
  const float* in_b    = (const float*)d_in[7];
  const float* out_w   = (const float*)d_in[8];
  const float* out_b   = (const float*)d_in[9];
  const float* W_s1    = (const float*)d_in[10];
  const float* b_s1    = (const float*)d_in[11];
  const float* W_s2    = (const float*)d_in[12];
  const float* b_s2    = (const float*)d_in[13];

  // ---- workspace layout (regions reused across phases) ----
  char* ws = (char*)d_ws;
  unsigned short* idx_bf = (unsigned short*)ws;                        // 102.4 MB, later kh/vh
  size_t off = (size_t)NIDX * DM * 2;                                  // 102,400,000
  size_t poolB = (size_t)BQ * NSEG * KTOP * sizeof(float);             // 25,690,112
  float* pool_s = (float*)(ws + off);
  int*   pool_i = (int*)(ws + off + poolB);
  size_t off2 = off + 2 * poolB;
  unsigned short* cand_bf = (unsigned short*)(ws + off2);              // 33.5 MB
  char* p = ws + off2 + (size_t)BQ * KTOP * DM * 2;
  unsigned short* query_bf = (unsigned short*)p; p += (size_t)BQ * DM * 2;
  unsigned short* Wcq_b    = (unsigned short*)p; p += (size_t)DM * DM * 2;
  unsigned short* Wcd_b    = (unsigned short*)p; p += (size_t)DM * DM * 2;
  unsigned short* Winp_b   = (unsigned short*)p; p += (size_t)3 * DM * DM * 2;
  unsigned short* Wout_b   = (unsigned short*)p; p += (size_t)DM * DM * 2;
  unsigned short* Ws1_b    = (unsigned short*)p; p += (size_t)(DM / 2) * DM * 2;
  unsigned short* q_bf     = (unsigned short*)p; p += (size_t)BQ * DM * 2;
  float*          qh_f     = (float*)p;          p += (size_t)BQ * DM * 4;
  unsigned short* ctx_bf   = (unsigned short*)p; p += (size_t)BQ * DM * 2;
  unsigned short* cross_bf = (unsigned short*)p; p += (size_t)BQ * DM * 2;
  unsigned short* h1_bf    = (unsigned short*)p; p += (size_t)BQ * (DM / 2) * 2;
  // region reuse after phase boundaries:
  unsigned short* dcand_bf = (unsigned short*)(ws + off);              // overlays pool (post-merge)
  unsigned short* kh_bf    = (unsigned short*)ws;                      // overlays idx_bf (post-retrieval)
  unsigned short* vh_bf    = (unsigned short*)(ws + (size_t)BQ * KTOP * DM * 2);

  int* out_idx = (int*)d_out;
  float* out_sc = (float*)d_out + (size_t)BQ * KTOP;
  float* out_rr = (float*)d_out + (size_t)2 * BQ * KTOP;

  // ---- phase 0: fp32 -> bf16 staging (index fits in 192MB L2 as bf16) ----
  k_f32_to_bf16<<<2048, 256, 0, stream>>>(index, idx_bf, (size_t)NIDX * DM);
  k_f32_to_bf16<<<128, 256, 0, stream>>>(query, query_bf, (size_t)BQ * DM);
  k_f32_to_bf16<<<64, 256, 0, stream>>>(W_cq, Wcq_b, (size_t)DM * DM);
  k_f32_to_bf16<<<64, 256, 0, stream>>>(W_cd, Wcd_b, (size_t)DM * DM);
  k_f32_to_bf16<<<128, 256, 0, stream>>>(in_w, Winp_b, (size_t)3 * DM * DM);
  k_f32_to_bf16<<<64, 256, 0, stream>>>(out_w, Wout_b, (size_t)DM * DM);
  k_f32_to_bf16<<<32, 256, 0, stream>>>(W_s1, Ws1_b, (size_t)(DM / 2) * DM);

  // ---- phase 1: retrieval (WMMA bf16 scores + hierarchical top-k) ----
  {
    dim3 g(BQ / 16, NSEG);
    size_t shm = (size_t)16 * SEG * 4 + 256 * 8;                       // 133,120 B < 160KB
    k_scores_topk<<<g, 256, shm, stream>>>(query_bf, idx_bf, pool_s, pool_i);
  }
  {
    size_t shm = (size_t)NSEG * KTOP * 4 + 256 * 8;                    // ~52 KB
    k_merge<<<BQ, 256, shm, stream>>>(pool_s, pool_i, out_idx, out_sc);
  }

  // ---- phase 2: rerank pipeline (all matmuls via WMMA bf16 GEMM) ----
  k_gather<<<BQ, 256, 0, stream>>>(out_idx, index, cand_bf);
  dim3 g1(BQ / 64, DM / 64);                                           // 8 x 4
  dim3 g2(BQ * KTOP / 64, DM / 64);                                    // 1024 x 4
  dim3 g3(BQ / 64, (DM / 2) / 64);                                     // 8 x 2
  // q = query @ W_cq^T + b_cq  (bf16 out)
  k_gemm_bf16<<<g1, 256, 0, stream>>>(query_bf, Wcq_b, b_cq, q_bf, BQ, DM, DM, 0, 1);
  // qh = q @ Wq^T + bq  (f32 out, consumed by attention)
  k_gemm_bf16<<<g1, 256, 0, stream>>>(q_bf, Winp_b, in_b, qh_f, BQ, DM, DM, 0, 0);
  // dcand = cand @ W_cd^T + b_cd
  k_gemm_bf16<<<g2, 256, 0, stream>>>(cand_bf, Wcd_b, b_cd, dcand_bf, BQ * KTOP, DM, DM, 0, 1);
  // kh = dcand @ Wk^T + bk ; vh = dcand @ Wv^T + bv
  k_gemm_bf16<<<g2, 256, 0, stream>>>(dcand_bf, Winp_b + (size_t)DM * DM, in_b + DM,
                                      kh_bf, BQ * KTOP, DM, DM, 0, 1);
  k_gemm_bf16<<<g2, 256, 0, stream>>>(dcand_bf, Winp_b + (size_t)2 * DM * DM, in_b + 2 * DM,
                                      vh_bf, BQ * KTOP, DM, DM, 0, 1);
  // attention -> ctx
  k_attn<<<BQ, 256, 0, stream>>>(qh_f, kh_bf, vh_bf, ctx_bf);
  // cross_out = ctx @ out_proj^T + b
  k_gemm_bf16<<<g1, 256, 0, stream>>>(ctx_bf, Wout_b, out_b, cross_bf, BQ, DM, DM, 0, 1);
  // h1 = relu(cross @ W_s1^T + b_s1)
  k_gemm_bf16<<<g3, 256, 0, stream>>>(cross_bf, Ws1_b, b_s1, h1_bf, BQ, DM / 2, DM, 1, 1);
  // rerank scores
  k_final<<<BQ / 8, 256, 0, stream>>>(h1_bf, W_s2, b_s2, out_rr);
}
